// ExpertizedLinear_1760936591663
// MI455X (gfx1250) — matrix operand order repaired and checked by reading, not compiled, not run on hardware
//
#include <hip/hip_runtime.h>

#define IN_F  2048
#define OUT_F 2048
#define NEXP  8
#define RANK  128
#define XPITCH 2056   // 2048 + 8 bf16 pad -> row stride 4112B, kills same-bank conflicts
#define HPITCH 136    // 128 + 8 bf16 pad

typedef __attribute__((ext_vector_type(16))) __bf16 v16bf;
typedef __attribute__((ext_vector_type(8)))  float  v8f;

union Frag32B { uint4 u[2]; v16bf v; };

__device__ __forceinline__ unsigned short f2bf(float f) {
  union { float f; unsigned int u; } c; c.f = f;
  unsigned int u = c.u;
  return (unsigned short)((u + 0x7fffu + ((u >> 16) & 1u)) >> 16); // RNE
}

// ---------------- router: norm + logits + softmax-top2 -> combine[T][8] ----
__global__ void __launch_bounds__(256) moe_router(
    const float* __restrict__ x, const float* __restrict__ rw,
    float* __restrict__ combine, int T) {
  __shared__ float rw_lds[IN_F * NEXP];            // 64 KB
  const int tid = threadIdx.x;
  for (int i = tid; i < IN_F * NEXP / 4; i += 256)
    ((float4*)rw_lds)[i] = ((const float4*)rw)[i];
  __syncthreads();

  const int wave = tid >> 5, lane = tid & 31;
  const int t = blockIdx.x * 8 + wave;
  if (t >= T) return;
  const float* xr = x + (size_t)t * IN_F;

  float ss = 0.f;
  float dot[NEXP];
#pragma unroll
  for (int e = 0; e < NEXP; ++e) dot[e] = 0.f;

  for (int k = lane; k < IN_F; k += 32) {
    float xv = xr[k];
    ss += xv * xv;
    const float4* rp = (const float4*)(rw_lds + k * NEXP);
    float4 r0 = rp[0], r1 = rp[1];
    dot[0] += xv * r0.x; dot[1] += xv * r0.y; dot[2] += xv * r0.z; dot[3] += xv * r0.w;
    dot[4] += xv * r1.x; dot[5] += xv * r1.y; dot[6] += xv * r1.z; dot[7] += xv * r1.w;
  }
#pragma unroll
  for (int off = 16; off > 0; off >>= 1) {        // wave32 butterfly
    ss += __shfl_xor(ss, off);
#pragma unroll
    for (int e = 0; e < NEXP; ++e) dot[e] += __shfl_xor(dot[e], off);
  }
  if (lane == 0) {
    float inv = 1.f / fmaxf(sqrtf(ss), 1e-12f);
    float lg[NEXP];
#pragma unroll
    for (int e = 0; e < NEXP; ++e) lg[e] = dot[e] * inv;
    int e1 = 0;
#pragma unroll
    for (int e = 1; e < NEXP; ++e) if (lg[e] > lg[e1]) e1 = e;   // first max (jax tie order)
    int e2 = (e1 == 0) ? 1 : 0;
#pragma unroll
    for (int e = 0; e < NEXP; ++e) if (e != e1 && lg[e] > lg[e2]) e2 = e;
    float c1 = 1.f / (1.f + __expf(lg[e2] - lg[e1]));  // softmax denom cancels in renorm
    float c2 = 1.f - c1;
    float* co = combine + (size_t)t * NEXP;
#pragma unroll
    for (int e = 0; e < NEXP; ++e) co[e] = (e == e1) ? c1 : ((e == e2) ? c2 : 0.f);
  }
}

// -------- weight convert/transpose to K-contiguous bf16 ---------------------
__global__ void __launch_bounds__(256) cvt_wa(const float* __restrict__ w,
                                              unsigned short* __restrict__ o) {
  int idx = blockIdx.x * 256 + threadIdx.x;        // out [e][n:128][k:2048]
  int k = idx & (IN_F - 1);
  int n = (idx >> 11) & (RANK - 1);
  int e = idx >> 18;
  o[idx] = f2bf(w[((size_t)e * IN_F + k) * RANK + n]);
}
__global__ void __launch_bounds__(256) cvt_wb(const float* __restrict__ w,
                                              unsigned short* __restrict__ o) {
  int idx = blockIdx.x * 256 + threadIdx.x;        // out [e][n:2048][k:128]
  int k = idx & (RANK - 1);
  int n = (idx >> 7) & (OUT_F - 1);
  int e = idx >> 18;
  o[idx] = f2bf(w[((size_t)e * RANK + k) * OUT_F + n]);
}

// -------- fused expert GEMM: 16 tokens per workgroup, 8 waves ---------------
__global__ void __launch_bounds__(256) moe_gemm(
    const float* __restrict__ x, const unsigned short* __restrict__ wa,
    const unsigned short* __restrict__ wb, const float* __restrict__ combine,
    float* __restrict__ out) {
  extern __shared__ unsigned char smem[];
  unsigned short* x_lds = (unsigned short*)smem;            // [16][XPITCH] bf16
  unsigned short* h_lds = x_lds + 16 * XPITCH;              // [16][HPITCH] bf16
  float* c_lds = (float*)(h_lds + 16 * HPITCH);             // [16][8]

  const int tid  = threadIdx.x;
  const int wave = tid >> 5, lane = tid & 31;
  const int col  = lane & 15, half = lane >> 4;
  const int t0   = blockIdx.x * 16;

  // stage x tile fp32 -> bf16 (read once, reused by all 8 experts)
  for (int i = tid; i < 16 * (IN_F / 4); i += 256) {
    int m = i >> 9, c4 = i & 511;
    float4 v = ((const float4*)(x + (size_t)(t0 + m) * IN_F))[c4];
    unsigned short* p = x_lds + m * XPITCH + c4 * 4;
    p[0] = f2bf(v.x); p[1] = f2bf(v.y); p[2] = f2bf(v.z); p[3] = f2bf(v.w);
  }
  if (tid < 128) c_lds[tid] = combine[(size_t)t0 * NEXP + tid];
  __syncthreads();

  const v8f vzero = {0.f, 0.f, 0.f, 0.f, 0.f, 0.f, 0.f, 0.f};
  v8f oacc[16];
#pragma unroll
  for (int j = 0; j < 16; ++j) oacc[j] = vzero;

  const unsigned short* arow = x_lds + col * XPITCH + half * 8;  // A: M=col

  for (int e = 0; e < NEXP; ++e) {
    // ---- phase 1: H[:, 16w..16w+15] = x_tile @ Wa[e] (K = 2048) ----
    v8f hacc = vzero;
    const unsigned short* bbase =
        wa + ((size_t)(e * RANK + wave * 16 + col)) * IN_F + half * 8;
#pragma unroll 4
    for (int ks = 0; ks < IN_F / 32; ++ks) {
      Frag32B a, b;
      const unsigned short* ap = arow + ks * 32;
      a.u[0] = *(const uint4*)ap;            // K: kb+8h .. +7
      a.u[1] = *(const uint4*)(ap + 16);     // K: kb+16+8h .. +7
      const unsigned short* bp = bbase + ks * 32;
      b.u[0] = *(const uint4*)bp;            // K: kb+8h .. +7
      b.u[1] = *(const uint4*)(bp + 16);     // K: kb+16+8h .. +7
      hacc = __builtin_amdgcn_wmma_f32_16x16x32_bf16(
          false, a.v, false, b.v, (short)0, hacc, false, false);
    }
    __syncthreads();   // prior phase-2 readers done with h_lds
    // fold routing weight into H rows while spilling to LDS (bf16)
#pragma unroll
    for (int r = 0; r < 8; ++r) {
      int m = r + half * 8;                       // C layout: M = r + 8*half
      float cw = c_lds[m * NEXP + e];
      h_lds[m * HPITCH + wave * 16 + col] = f2bf(hacc[r] * cw);
    }
    __syncthreads();
    // ---- phase 2: out[:, 256w..+255] += Hs @ Wb[e] (K = 128) ----
    Frag32B a2[4];
#pragma unroll
    for (int s = 0; s < 4; ++s) {
      const unsigned short* ap = h_lds + col * HPITCH + s * 32 + half * 8;
      a2[s].u[0] = *(const uint4*)ap;
      a2[s].u[1] = *(const uint4*)(ap + 16);
    }
#pragma unroll
    for (int j = 0; j < 16; ++j) {
      const unsigned short* bb =
          wb + ((size_t)(e * OUT_F + wave * 256 + j * 16 + col)) * RANK + half * 8;
#pragma unroll
      for (int s = 0; s < 4; ++s) {
        Frag32B b;
        const unsigned short* bp = bb + s * 32;
        b.u[0] = *(const uint4*)bp;
        b.u[1] = *(const uint4*)(bp + 16);
        oacc[j] = __builtin_amdgcn_wmma_f32_16x16x32_bf16(
            false, a2[s].v, false, b.v, (short)0, oacc[j], false, false);
      }
    }
  }

  // epilogue: fp32 store, row M = r + 8*half, col N = 256w + 16j + col
#pragma unroll
  for (int j = 0; j < 16; ++j) {
    float* op = out + (size_t)(t0 + half * 8) * OUT_F + wave * 256 + j * 16 + col;
#pragma unroll
    for (int r = 0; r < 8; ++r) op[(size_t)r * OUT_F] = oacc[j][r];
  }
}

extern "C" void kernel_launch(void* const* d_in, const int* in_sizes, int n_in,
                              void* d_out, int out_size, void* d_ws, size_t ws_size,
                              hipStream_t stream) {
  const float* x  = (const float*)d_in[0];
  const float* rw = (const float*)d_in[1];
  const float* Wa = (const float*)d_in[2];
  const float* Wb = (const float*)d_in[3];
  float* out = (float*)d_out;
  const int T = in_sizes[0] / IN_F;                  // 16384 tokens

  // workspace: combine[T][8] f32 | Wa_t bf16 | Wb_t bf16  (~8.9 MB total)
  float* combine = (float*)d_ws;
  unsigned short* wa_t =
      (unsigned short*)((char*)d_ws + (size_t)T * NEXP * sizeof(float));
  unsigned short* wb_t = wa_t + (size_t)NEXP * RANK * IN_F;

  cvt_wa<<<(NEXP * RANK * IN_F) / 256, 256, 0, stream>>>(Wa, wa_t);
  cvt_wb<<<(NEXP * RANK * OUT_F) / 256, 256, 0, stream>>>(Wb, wb_t);
  moe_router<<<(T + 7) / 8, 256, 0, stream>>>(x, rw, combine, T);

  size_t smem = (size_t)(16 * XPITCH + 16 * HPITCH) * sizeof(unsigned short) +
                16 * NEXP * sizeof(float);           // ~70.7 KB of 320 KB/WGP
  moe_gemm<<<T / 16, 256, smem, stream>>>(x, wa_t, wb_t, combine, out);
}
